// Lstm_AutoEncoder_68813966017024
// MI455X (gfx1250) — compile-verified, matrix operand order checked by reading, not run on hardware
//
#include <hip/hip_runtime.h>

// MI455X / gfx1250 persistent-LSTM autoencoder.
// B=128, T=1024, X=64, H=256, gates 4H=1024. Gate order i,f,g,o.
//
// ws layout (bytes), all f16 weight copies pre-transposed to [N][K] (K-major):
constexpr size_t OFF_UET = 0;                       // Ue^T  [1024][256] f16 512KB
constexpr size_t OFF_UDT = OFF_UET + 1024*256*2;    // Ud^T  [1024][256] f16 512KB
constexpr size_t OFF_WDT = OFF_UDT + 1024*256*2;    // Wd^T  [1024][256] f16 512KB
constexpr size_t OFF_WET = OFF_WDT + 1024*256*2;    // We^T  [1024][64]  f16 128KB
constexpr size_t OFF_WOT = OFF_WET + 1024*64*2;     // Wo^T  [64][256]   f16 32KB
constexpr size_t OFF_H0  = OFF_WOT + 64*256*2;      // h ping [128][256] f16 64KB
constexpr size_t OFF_H1  = OFF_H0 + 128*256*2;      // h pong [128][256] f16 64KB
constexpr size_t OFF_CNT = OFF_H1 + 128*256*2;      // grid-barrier counter (256B pad)
constexpr size_t OFF_HS  = OFF_CNT + 256;           // hs_dec [1024][128][256] f16 64MB
// total ~66MB of d_ws

#define NBLK 16
#define GAS __attribute__((address_space(1)))       // force global_load/store path

typedef __attribute__((ext_vector_type(16))) _Float16 v16h;
typedef __attribute__((ext_vector_type(8)))  _Float16 v8h;
typedef __attribute__((ext_vector_type(8)))  float    v8f;
typedef __attribute__((ext_vector_type(4)))  unsigned uint4v;

__device__ __forceinline__ v8f wmma_f16(v16h a, v16h b, v8f c) {
  return __builtin_amdgcn_wmma_f32_16x16x32_f16(false, a, false, b, (short)0, c,
                                                false, false);
}

// ---- fragment loaders for 16x16x32 f16 WMMA from a K-major tile ----
// t0 = &src[row0*ld + k0]; per-lane row = lane&15, K offset 8*(lane>>4).
// e[0..7] = K 8*half..+8 ; e[8..15] = K 16+8*half..+8  (two b128 loads)

// LDS source (generic pointer derived from shared -> ds_load_b128)
__device__ __forceinline__ v16h load_tile_l(const _Float16* t0, int ld, int lane) {
  const _Float16* p = t0 + (lane & 15) * ld + ((lane >> 4) << 3);
  v8h lo = *(const v8h*)(p);
  v8h hi = *(const v8h*)(p + 16);
  v16h r;
#pragma unroll
  for (int i = 0; i < 8; ++i) { r[i] = lo[i]; r[8 + i] = hi[i]; }
  return r;
}

// Global f16 source -> global_load_b128
__device__ __forceinline__ v16h load_tile_g(const _Float16* t0, int ld, int lane) {
  GAS const _Float16* p =
      (GAS const _Float16*)t0 + (lane & 15) * ld + ((lane >> 4) << 3);
  v8h lo = *(GAS const v8h*)(p);
  v8h hi = *(GAS const v8h*)(p + 16);
  v16h r;
#pragma unroll
  for (int i = 0; i < 8; ++i) { r[i] = lo[i]; r[8 + i] = hi[i]; }
  return r;
}

// Global f32 source (for x), converted on the fly
__device__ __forceinline__ v16h load_tile_gf32(const float* t0, int ld, int lane) {
  GAS const float* p = (GAS const float*)t0 + (lane & 15) * ld + ((lane >> 4) << 3);
  v8f lo = *(GAS const v8f*)(p);
  v8f hi = *(GAS const v8f*)(p + 16);
  v16h r;
#pragma unroll
  for (int i = 0; i < 8; ++i) { r[i] = (_Float16)lo[i]; r[8 + i] = (_Float16)hi[i]; }
  return r;
}

__device__ __forceinline__ void store_h_g(_Float16* p, _Float16 v) {
  *(GAS _Float16*)p = v;
}
__device__ __forceinline__ void store_f_g(float* p, float v) {
  *(GAS float*)p = v;
}

__device__ __forceinline__ float sigm(float v) { return 1.0f / (1.0f + __expf(-v)); }

// Software grid barrier: monotonic counter in L2, agent-scope fences.
__device__ __forceinline__ void grid_sync(unsigned* cnt, unsigned need) {
  __syncthreads();
  if (threadIdx.x == 0) {
    __builtin_amdgcn_fence(__ATOMIC_RELEASE, "agent");
    __hip_atomic_fetch_add(cnt, 1u, __ATOMIC_RELAXED, __HIP_MEMORY_SCOPE_AGENT);
    while (__hip_atomic_load(cnt, __ATOMIC_RELAXED, __HIP_MEMORY_SCOPE_AGENT) < need)
      __builtin_amdgcn_s_sleep(1);
    __builtin_amdgcn_fence(__ATOMIC_ACQUIRE, "agent");
  }
  __syncthreads();
}

// Convert+transpose: src f32 [K][N] -> dst f16 [N][K].
__global__ void k_cvt_t(const float* __restrict__ src, _Float16* __restrict__ dst,
                        int K, int N) {
  int i = blockIdx.x * blockDim.x + threadIdx.x;
  if (i >= N * K) return;
  int n = i / K, k = i - n * K;
  store_h_g(dst + i, (_Float16)src[k * N + n]);
}

__global__ void k_zero(unsigned* __restrict__ p, int n) {
  int i = blockIdx.x * blockDim.x + threadIdx.x;
  if (i < n) *(GAS unsigned*)(p + i) = 0u;
}

// Persistent recurrent kernel: 16 workgroups x 256 threads (8 waves).
// WG j owns h-columns [16j,16j+16) => z-columns {g*256+16j+nl}.
// Wave w owns batch rows [16w,16w+16). c tile lives in registers (v8f fragment).
__global__ void __launch_bounds__(256, 1)
k_lstm(const float* __restrict__ x, const float* __restrict__ be,
       const float* __restrict__ bd, char* __restrict__ ws) {
  extern __shared__ __align__(16) char smem[];
  _Float16* lds0 = (_Float16*)smem;             // 64x256 f16: Ue slice -> Ud slice
  _Float16* lds1 = (_Float16*)(smem + 32768);   // We slice (64x64) -> Wd slice (64x256)

  const _Float16* Uet = (const _Float16*)(ws + OFF_UET);
  const _Float16* Udt = (const _Float16*)(ws + OFF_UDT);
  const _Float16* Wdt = (const _Float16*)(ws + OFF_WDT);
  const _Float16* Wet = (const _Float16*)(ws + OFF_WET);
  _Float16* h0 = (_Float16*)(ws + OFF_H0);
  _Float16* h1 = (_Float16*)(ws + OFF_H1);
  _Float16* hs = (_Float16*)(ws + OFF_HS);
  unsigned* cnt = (unsigned*)(ws + OFF_CNT);

  const int tid  = threadIdx.x;
  const int lane = tid & 31;
  const int wv   = tid >> 5;
  const int m0   = wv * 16;              // batch-row tile
  const int c0   = blockIdx.x * 16;      // h-column slice
  const int nl   = lane & 15;
  const int half = lane >> 4;
  const int hbase = m0 * 256;

  // --- stage encoder weight slices into LDS (persist across the whole phase) ---
  for (int i = tid; i < 64 * 32; i += 256) {          // Ue^T slice: 64 rows x 512B
    int row = i >> 5, q = i & 31;
    int srow = (row >> 4) * 256 + c0 + (row & 15);    // z-col = g*256 + c0 + nl
    ((uint4v*)(lds0 + row * 256))[q] = ((GAS const uint4v*)(Uet + srow * 256))[q];
  }
  for (int i = tid; i < 64 * 8; i += 256) {           // We^T slice: 64 rows x 128B
    int row = i >> 3, q = i & 7;
    int srow = (row >> 4) * 256 + c0 + (row & 15);
    ((uint4v*)(lds1 + row * 64))[q] = ((GAS const uint4v*)(Wet + srow * 64))[q];
  }
  __syncthreads();

  float beg[4];
#pragma unroll
  for (int g = 0; g < 4; ++g) beg[g] = be[g * 256 + c0 + nl];

  int bar = 0;
  v8f cr = {};   // cell-state tile, lives in registers for the whole phase

  // ===================== encoder: 1024 steps =====================
  for (int t = 0; t < 1024; ++t) {
    v8f acc[4];
    {
      v8f z = {};
#pragma unroll
      for (int g = 0; g < 4; ++g) acc[g] = z;   // first WMMA chains from zeroed C
    }

    // x_t @ We  (K = 64, two chunks), prefetch next step's x rows
    const float* xt = x + m0 * 65536 + t * 64;
    __builtin_prefetch(xt + 64, 0, 1);
#pragma unroll
    for (int kc = 0; kc < 2; ++kc) {
      int k0 = kc * 32;
      v16h a = load_tile_gf32(xt + k0, 65536, lane);
#pragma unroll
      for (int g = 0; g < 4; ++g) {
        v16h b = load_tile_l(lds1 + (g * 16) * 64 + k0, 64, lane);
        acc[g] = wmma_f16(a, b, acc[g]);
      }
    }

    // h_{t-1} @ Ue  (K = 256, eight chunks), software-pipelined one chunk ahead
    const _Float16* hr = (t & 1) ? h0 : h1;
    v16h a = load_tile_g(hr + hbase, 256, lane);
    v16h bb[4];
#pragma unroll
    for (int g = 0; g < 4; ++g) bb[g] = load_tile_l(lds0 + (g * 16) * 256, 256, lane);
#pragma unroll
    for (int kc = 0; kc < 8; ++kc) {
      v16h an = a;
      v16h bn[4];
#pragma unroll
      for (int g = 0; g < 4; ++g) bn[g] = bb[g];
      if (kc < 7) {
        int k1 = (kc + 1) * 32;
        an = load_tile_g(hr + hbase + k1, 256, lane);
#pragma unroll
        for (int g = 0; g < 4; ++g)
          bn[g] = load_tile_l(lds0 + (g * 16) * 256 + k1, 256, lane);
      }
#pragma unroll
      for (int g = 0; g < 4; ++g) acc[g] = wmma_f16(a, bb[g], acc[g]);
      a = an;
#pragma unroll
      for (int g = 0; g < 4; ++g) bb[g] = bn[g];
    }

    // gates + state update (bias folded here instead of acc init)
    v8f h8;
#pragma unroll
    for (int e = 0; e < 8; ++e) {
      float iv = sigm(acc[0][e] + beg[0]);
      float fv = sigm(acc[1][e] + beg[1]);
      float gv = fmaxf(acc[2][e] + beg[2], 0.0f);
      float ov = sigm(acc[3][e] + beg[3]);
      float cv = fv * cr[e] + iv * gv;
      cr[e] = cv;
      h8[e] = ov * fmaxf(cv, 0.0f);
    }
    _Float16* hw = (t & 1) ? h1 : h0;
    int col = c0 + nl;
#pragma unroll
    for (int e = 0; e < 8; ++e)
      store_h_g(hw + (m0 + e + 8 * half) * 256 + col, (_Float16)h8[e]);

    grid_sync(cnt, (unsigned)(++bar) * NBLK);
  }

  // --- swap weight slices: Ud -> lds0, Wd -> lds1 ---
  for (int i = tid; i < 64 * 32; i += 256) {
    int row = i >> 5, q = i & 31;
    int srow = (row >> 4) * 256 + c0 + (row & 15);
    ((uint4v*)(lds0 + row * 256))[q] = ((GAS const uint4v*)(Udt + srow * 256))[q];
  }
  for (int i = tid; i < 64 * 32; i += 256) {
    int row = i >> 5, q = i & 31;
    int srow = (row >> 4) * 256 + c0 + (row & 15);
    ((uint4v*)(lds1 + row * 256))[q] = ((GAS const uint4v*)(Wdt + srow * 256))[q];
  }
  __syncthreads();

  // zd = h_enc @ Wd + bd  (constant across all decoder steps -> keep in regs)
  v8f zd[4];
  {
    v8f z = {};
#pragma unroll
    for (int g = 0; g < 4; ++g) zd[g] = z;
  }
#pragma unroll
  for (int kc = 0; kc < 8; ++kc) {
    int k0 = kc * 32;
    v16h a = load_tile_g(h1 + hbase + k0, 256, lane);   // h(1023) is in h1
#pragma unroll
    for (int g = 0; g < 4; ++g) {
      v16h b = load_tile_l(lds1 + (g * 16) * 256 + k0, 256, lane);
      zd[g] = wmma_f16(a, b, zd[g]);
    }
  }
#pragma unroll
  for (int g = 0; g < 4; ++g) {
    float bb = bd[g * 256 + c0 + nl];
#pragma unroll
    for (int e = 0; e < 8; ++e) zd[g][e] += bb;
  }
  grid_sync(cnt, (unsigned)(++bar) * NBLK);   // everyone done reading h_enc

  // zero own column-slice of h1 so the decoder starts from h = 0
  {
    int row = tid >> 1;
    int o8  = (tid & 1) * 8;
    v8h z = {};
    *(GAS v8h*)((GAS _Float16*)h1 + row * 256 + c0 + o8) = z;
  }
  grid_sync(cnt, (unsigned)(++bar) * NBLK);

#pragma unroll
  for (int e = 0; e < 8; ++e) cr[e] = 0.0f;   // reset cell state

  // ===================== decoder: 1024 steps =====================
  for (int s = 0; s < 1024; ++s) {
    v8f acc[4];
    {
      v8f z = {};
#pragma unroll
      for (int g = 0; g < 4; ++g) acc[g] = z;
    }
    const _Float16* hr = (s & 1) ? h0 : h1;
    v16h a = load_tile_g(hr + hbase, 256, lane);
    v16h bb[4];
#pragma unroll
    for (int g = 0; g < 4; ++g) bb[g] = load_tile_l(lds0 + (g * 16) * 256, 256, lane);
#pragma unroll
    for (int kc = 0; kc < 8; ++kc) {
      v16h an = a;
      v16h bn[4];
#pragma unroll
      for (int g = 0; g < 4; ++g) bn[g] = bb[g];
      if (kc < 7) {
        int k1 = (kc + 1) * 32;
        an = load_tile_g(hr + hbase + k1, 256, lane);
#pragma unroll
        for (int g = 0; g < 4; ++g)
          bn[g] = load_tile_l(lds0 + (g * 16) * 256 + k1, 256, lane);
      }
#pragma unroll
      for (int g = 0; g < 4; ++g) acc[g] = wmma_f16(a, bb[g], acc[g]);
      a = an;
#pragma unroll
      for (int g = 0; g < 4; ++g) bb[g] = bn[g];
    }

    // gates; zd (h_enc@Wd + bd) folded here, per-element
    v8f h8;
#pragma unroll
    for (int e = 0; e < 8; ++e) {
      float iv = sigm(acc[0][e] + zd[0][e]);
      float fv = sigm(acc[1][e] + zd[1][e]);
      float gv = fmaxf(acc[2][e] + zd[2][e], 0.0f);
      float ov = sigm(acc[3][e] + zd[3][e]);
      float cv = fv * cr[e] + iv * gv;
      cr[e] = cv;
      h8[e] = ov * fmaxf(cv, 0.0f);
    }
    _Float16* hw = (s & 1) ? h1 : h0;
    int col = c0 + nl;
#pragma unroll
    for (int e = 0; e < 8; ++e) {
      _Float16 hv = (_Float16)h8[e];
      int row = m0 + e + 8 * half;
      store_h_g(hw + row * 256 + col, hv);
      store_h_g(hs + (s * 128 + row) * 256 + col, hv);  // record for projection
    }
    grid_sync(cnt, (unsigned)(++bar) * NBLK);
  }
}

// Output projection: out[b][t][:] = hs_dec[t][b][:] @ Wo + bo.
// GEMM M=131072, N=64, K=256 — fully parallel, HBM-bound (~96MB traffic).
__global__ void __launch_bounds__(256)
k_out(const _Float16* __restrict__ hsd, const _Float16* __restrict__ wot,
      const float* __restrict__ bo, float* __restrict__ out) {
  int tid  = threadIdx.x;
  int lane = tid & 31;
  int w    = blockIdx.x * 8 + (tid >> 5);   // global wave id, one M-tile each
  int m0   = w * 16;
  int nl   = lane & 15;
  int half = lane >> 4;

  v8f acc[4];
  {
    v8f z = {};
#pragma unroll
    for (int nt = 0; nt < 4; ++nt) acc[nt] = z;
  }
  v16h a = load_tile_g(hsd + m0 * 256, 256, lane);
  v16h bb[4];
#pragma unroll
  for (int nt = 0; nt < 4; ++nt) bb[nt] = load_tile_g(wot + (nt * 16) * 256, 256, lane);
#pragma unroll
  for (int kc = 0; kc < 8; ++kc) {
    v16h an = a;
    v16h bn[4];
#pragma unroll
    for (int nt = 0; nt < 4; ++nt) bn[nt] = bb[nt];
    if (kc < 7) {
      int k1 = (kc + 1) * 32;
      an = load_tile_g(hsd + m0 * 256 + k1, 256, lane);
#pragma unroll
      for (int nt = 0; nt < 4; ++nt)
        bn[nt] = load_tile_g(wot + (nt * 16) * 256 + k1, 256, lane);
    }
#pragma unroll
    for (int nt = 0; nt < 4; ++nt) acc[nt] = wmma_f16(a, bb[nt], acc[nt]);
    a = an;
#pragma unroll
    for (int nt = 0; nt < 4; ++nt) bb[nt] = bn[nt];
  }
#pragma unroll
  for (int nt = 0; nt < 4; ++nt) {
    float bias = bo[nt * 16 + nl];
#pragma unroll
    for (int e = 0; e < 8; ++e) {
      int r = m0 + e + 8 * half;          // r = t*128 + b
      int bb2 = r & 127;
      int tt  = r >> 7;
      store_f_g(out + (bb2 * 1024 + tt) * 64 + nt * 16 + nl, acc[nt][e] + bias);
    }
  }
}

extern "C" void kernel_launch(void* const* d_in, const int* in_sizes, int n_in,
                              void* d_out, int out_size, void* d_ws, size_t ws_size,
                              hipStream_t stream) {
  (void)in_sizes; (void)n_in; (void)out_size; (void)ws_size;
  const float* x  = (const float*)d_in[0];
  const float* We = (const float*)d_in[1];
  const float* Ue = (const float*)d_in[2];
  const float* be = (const float*)d_in[3];
  const float* Wd = (const float*)d_in[4];
  const float* Ud = (const float*)d_in[5];
  const float* bd = (const float*)d_in[6];
  const float* Wo = (const float*)d_in[7];
  const float* bo = (const float*)d_in[8];
  char*  ws  = (char*)d_ws;
  float* out = (float*)d_out;

  // Weight prep: convert f32 -> f16 and transpose to K-major (N-row) layout.
  k_cvt_t<<<(1024 * 256 + 255) / 256, 256, 0, stream>>>(Ue, (_Float16*)(ws + OFF_UET), 256, 1024);
  k_cvt_t<<<(1024 * 256 + 255) / 256, 256, 0, stream>>>(Ud, (_Float16*)(ws + OFF_UDT), 256, 1024);
  k_cvt_t<<<(1024 * 256 + 255) / 256, 256, 0, stream>>>(Wd, (_Float16*)(ws + OFF_WDT), 256, 1024);
  k_cvt_t<<<(1024 * 64  + 255) / 256, 256, 0, stream>>>(We, (_Float16*)(ws + OFF_WET), 64, 1024);
  k_cvt_t<<<(64 * 256   + 255) / 256, 256, 0, stream>>>(Wo, (_Float16*)(ws + OFF_WOT), 256, 64);
  // Zero h ping-pong buffers + barrier counter (re-done every call -> graph-safe).
  k_zero<<<(32832 + 255) / 256, 256, 0, stream>>>((unsigned*)(ws + OFF_H0), 32832);

  // Persistent recurrent kernel: 16 WGs x 256 threads, 64KB dynamic LDS each.
  k_lstm<<<NBLK, 256, 65536, stream>>>(x, be, bd, ws);

  // Final TimeDistributed Dense: 8192 M-tiles, 8 waves/block.
  k_out<<<1024, 256, 0, stream>>>((const _Float16*)(ws + OFF_HS),
                                  (const _Float16*)(ws + OFF_WOT), bo, out);
}
